// MultiHeadAttention_18313740550365
// MI455X (gfx1250) — matrix operand order repaired and checked by reading, not compiled, not run on hardware
//
#include <hip/hip_runtime.h>

// ---------------- types ----------------
typedef __attribute__((ext_vector_type(16))) __bf16        bf16x16;
typedef __attribute__((ext_vector_type(8)))  float         f32x8;
typedef __attribute__((ext_vector_type(4)))  float         f32x4;
typedef __attribute__((ext_vector_type(8)))  unsigned int  u32x8;
typedef __attribute__((ext_vector_type(4)))  unsigned int  u32x4;
typedef __attribute__((ext_vector_type(2)))  unsigned int  u32x2;

#define DM 1024   // d_model
#define NH 16     // heads
#define HD 64     // head dim
#define BB 2      // batch
#define SS 2048   // seq
#define MROWS (BB*SS)  // 4096 token rows

// Q pre-scale: (1/sqrt(64)) * log2(e)  -> softmax done in exp2 domain
#define QSCALE 0.180336880111120419f

// ---------------- helpers ----------------
__device__ __forceinline__ unsigned short bfbits(float f) {
  __bf16 h = (__bf16)f;                       // native v_cvt, RNE
  return __builtin_bit_cast(unsigned short, h);
}
// fragment from two 16-byte chunks
__device__ __forceinline__ bf16x16 frag_ld2(const __bf16* p0, const __bf16* p1) {
  u32x4 lo = *(const u32x4*)p0;
  u32x4 hi = *(const u32x4*)p1;
  u32x8 v;
  v[0]=lo[0]; v[1]=lo[1]; v[2]=lo[2]; v[3]=lo[3];
  v[4]=hi[0]; v[5]=hi[1]; v[6]=hi[2]; v[7]=hi[3];
  return __builtin_bit_cast(bf16x16, v);
}
// B-operand fragment: 32 contiguous bytes (global, 32B aligned)
__device__ __forceinline__ bf16x16 frag_ld(const __bf16* p) {
  return *(const bf16x16*)p;
}
// B-operand fragment from LDS: two 16B chunks 32 bytes apart, 16B alignment
__device__ __forceinline__ bf16x16 frag_lds(const __bf16* p) {
  return frag_ld2(p, p + 16);
}
__device__ __forceinline__ f32x8 wmma_bf16(bf16x16 a, bf16x16 b, f32x8 c) {
  return __builtin_amdgcn_wmma_f32_16x16x32_bf16(false, a, false, b, (short)0, c,
                                                 false, false);
}
__device__ __forceinline__ f32x8 zero8() {
  f32x8 z;
#pragma unroll
  for (int i = 0; i < 8; ++i) z[i] = 0.0f;
  return z;
}

// DPP16 ROW_XMASK butterfly within 16-lane rows: pure VALU, no DS traffic.
template <int CTRL>
__device__ __forceinline__ float dppmov(float v) {
  return __builtin_bit_cast(
      float, __builtin_amdgcn_update_dpp(0, __builtin_bit_cast(int, v), CTRL,
                                         0xF, 0xF, true));
}
__device__ __forceinline__ float bfly_max16(float v) {
  v = fmaxf(v, dppmov<0x161>(v));  // ROW_XMASK1
  v = fmaxf(v, dppmov<0x162>(v));  // ROW_XMASK2
  v = fmaxf(v, dppmov<0x164>(v));  // ROW_XMASK4
  v = fmaxf(v, dppmov<0x168>(v));  // ROW_XMASK8
  return v;
}
__device__ __forceinline__ float bfly_add16(float v) {
  v += dppmov<0x161>(v);
  v += dppmov<0x162>(v);
  v += dppmov<0x164>(v);
  v += dppmov<0x168>(v);
  return v;
}

// Async global -> LDS copy of one 16-byte chunk (ASYNCcnt-tracked).
__device__ __forceinline__ void async_b128(unsigned lds_off, const void* gptr) {
  asm volatile("global_load_async_to_lds_b128 %0, %1, off"
               :: "v"(lds_off), "v"((unsigned long long)(uintptr_t)gptr)
               : "memory");
}
__device__ __forceinline__ void wait_async0() {
  asm volatile("s_wait_asynccnt 0x0" ::: "memory");
}

// ---------------- fp32 -> bf16 convert ----------------
__global__ void cvt_f32_bf16(const float* __restrict__ s, __bf16* __restrict__ d) {
  size_t i = ((size_t)blockIdx.x * blockDim.x + threadIdx.x) * 4;
  f32x4 v = *(const f32x4*)(s + i);
  u32x2 o;
  o[0] = (unsigned)bfbits(v[0]) | ((unsigned)bfbits(v[1]) << 16);
  o[1] = (unsigned)bfbits(v[2]) | ((unsigned)bfbits(v[3]) << 16);
  *(u32x2*)(d + i) = o;
}

// ---------------- projection GEMM: y = A @ W^T + bias ----------------
// A: [4096,1024] bf16 row-major.  W: [1024,1024] bf16 row-major (row n = B column n).
// MODE 0: Q -> obf [B,H,S,HD], scaled by QSCALE (softmax scale * log2e)
// MODE 1: K -> obf [B,H,S,HD]
// MODE 2: V -> obf [B,H,HD,S]  (transposed)
// MODE 3: O -> of32 [4096,1024]
template <int MODE>
__global__ __launch_bounds__(128) void proj_gemm(const __bf16* __restrict__ A,
                                                 const __bf16* __restrict__ W,
                                                 const float*  __restrict__ bias,
                                                 __bf16* __restrict__ obf,
                                                 float*  __restrict__ of32) {
  const int lane = threadIdx.x & 31;
  const int wid  = threadIdx.x >> 5;
  const int ln16 = lane & 15;
  const int half = lane >> 4;
  const int m0 = blockIdx.y * 128 + (wid >> 1) * 64;
  const int n0 = blockIdx.x * 128 + (wid & 1) * 64;

  f32x8 acc[4][4];
#pragma unroll
  for (int ti = 0; ti < 4; ++ti)
#pragma unroll
    for (int tj = 0; tj < 4; ++tj) acc[ti][tj] = zero8();

  const __bf16* Ap = A + (size_t)(m0 + ln16) * DM + half * 8;
  const __bf16* Wp = W + (size_t)(n0 + ln16) * DM + half * 8;

  for (int kk = 0; kk < DM; kk += 32) {
    bf16x16 af[4], wf[4];
#pragma unroll
    for (int t = 0; t < 4; ++t) {
      const __bf16* pa = Ap + (size_t)t * 16 * DM + kk;
      af[t] = frag_ld2(pa, pa + 16);
      wf[t] = frag_lds(Wp + (size_t)t * 16 * DM + kk);
    }
#pragma unroll
    for (int ti = 0; ti < 4; ++ti)
#pragma unroll
      for (int tj = 0; tj < 4; ++tj)
        acc[ti][tj] = wmma_bf16(af[ti], wf[tj], acc[ti][tj]);
  }

  // epilogue
#pragma unroll
  for (int tj = 0; tj < 4; ++tj) {
    const int gn = n0 + tj * 16 + ln16;  // output column
    const float bv = bias[gn];
#pragma unroll
    for (int ti = 0; ti < 4; ++ti) {
      const int gmb = m0 + ti * 16 + 8 * half;  // first of this lane's 8 rows
      float v[8];
#pragma unroll
      for (int r = 0; r < 8; ++r) {
        v[r] = acc[ti][tj][r] + bv;
        if (MODE == 0) v[r] *= QSCALE;
      }
      const int bhi  = (gmb >> 11) * NH + (gn >> 6);
      const int srow = gmb & (SS - 1);
      const int dcol = gn & (HD - 1);
      if (MODE == 0 || MODE == 1) {
#pragma unroll
        for (int r = 0; r < 8; ++r)
          obf[((size_t)bhi * SS + srow + r) * HD + dcol] = (__bf16)v[r];
      } else if (MODE == 2) {
        u32x4 pk;
#pragma unroll
        for (int i = 0; i < 4; ++i)
          pk[i] = (unsigned)bfbits(v[2 * i]) | ((unsigned)bfbits(v[2 * i + 1]) << 16);
        *(u32x4*)(obf + ((size_t)bhi * HD + dcol) * SS + srow) = pk;
      } else {
#pragma unroll
        for (int r = 0; r < 8; ++r)
          of32[(size_t)(gmb + r) * DM + gn] = v[r];
      }
    }
  }
}

// ---------------- flash attention (64-key blocks, async K/V staging) ----------------
// Q,K: [B,H,S,HD] bf16 (Q pre-scaled by QSCALE). Vt: [B,H,HD,S] bf16. ctx: [B*S,DM] bf16.
#define PSTR 72                 // LDS row stride (elements): 36 dwords, conflict-free
#define KVB  (64 * PSTR * 2)    // bytes per K or V buffer (9216)
__global__ __launch_bounds__(128) void attn_kernel(const __bf16* __restrict__ Qb,
                                                   const __bf16* __restrict__ Kb,
                                                   const __bf16* __restrict__ Vt,
                                                   const int*    __restrict__ mask,
                                                   __bf16* __restrict__ ctx) {
  const int tid  = threadIdx.x;
  const int lane = tid & 31;
  const int wid  = tid >> 5;
  const int ln16 = lane & 15;
  const int half = lane >> 4;
  const int b  = blockIdx.z;
  const int h  = blockIdx.y;
  const int q0 = blockIdx.x * 64 + wid * 16;
  const int bh = b * NH + h;

  const __bf16* Qh = Qb + ((size_t)bh * SS + q0) * HD;
  const __bf16* Kh = Kb + (size_t)bh * SS * HD;
  const __bf16* Vh = Vt + (size_t)bh * HD * SS;
  const int* mk = mask + b * SS;

  // double-buffered K/V tiles shared by all 4 waves + per-wave P tiles
  __shared__ alignas(16) __bf16 sK[2][64][PSTR];
  __shared__ alignas(16) __bf16 sV[2][64][PSTR];
  __shared__ alignas(16) __bf16 sP[4][16][PSTR];
  __bf16* P = &sP[wid][0][0];

  const unsigned ldsK = (unsigned)(uintptr_t)&sK[0][0][0];
  const unsigned ldsV = (unsigned)(uintptr_t)&sV[0][0][0];
  const int c8 = tid & 7;  // this thread's 16B column chunk

  // Q A-operand fragments (d 0..31 and 32..63), resident for the whole pass
  bf16x16 qf0, qf1;
  {
    const __bf16* p = Qh + (size_t)ln16 * HD + half * 8;
    qf0 = frag_ld2(p, p + 16);
    qf1 = frag_ld2(p + 32, p + 48);
  }

  float mrun[8], lrun[8];
#pragma unroll
  for (int r = 0; r < 8; ++r) { mrun[r] = -1e30f; lrun[r] = 0.0f; }
  f32x8 acc[4];
#pragma unroll
  for (int tj = 0; tj < 4; ++tj) acc[tj] = zero8();

  // ---- async stage of one 64-key block into buffer `buf` ----
  auto stage = [&](int kblk, int buf) {
#pragma unroll
    for (int i = 0; i < 4; ++i) {
      const int ch  = tid + i * 128;       // 0..511
      const int row = ch >> 3;
      const unsigned loff = (unsigned)buf * KVB + row * (PSTR * 2) + c8 * 16;
      async_b128(ldsK + loff, Kh + (size_t)(kblk + row) * HD + c8 * 8);
      async_b128(ldsV + loff, Vh + (size_t)row * SS + kblk + c8 * 8);
    }
  };

  // prologue: fill buffer 0
  stage(0, 0);
  wait_async0();
  __syncthreads();

  int buf = 0;
  for (int kblk = 0; kblk < SS; kblk += 64) {
    // prefetch next block into the other buffer (overlaps with compute)
    if (kblk + 64 < SS) stage(kblk + 64, buf ^ 1);

    // ---- load ALL K fragments first (distinct regs -> clauses + partial waits)
    bf16x16 kf[8];
#pragma unroll
    for (int t = 0; t < 4; ++t) {
      const __bf16* kp = &sK[buf][t * 16 + ln16][half * 8];
      kf[2 * t]     = frag_lds(kp);
      kf[2 * t + 1] = frag_lds(kp + 32);
    }
    // ---- scores: 16 queries x 64 keys (log2 domain) ----
    f32x8 sc[4];
#pragma unroll
    for (int t = 0; t < 4; ++t) {
      const float madd = (mk[kblk + t * 16 + ln16] == 0) ? -1.5e9f : 0.0f;
      f32x8 s = zero8();
      s = wmma_bf16(qf0, kf[2 * t], s);
      s = wmma_bf16(qf1, kf[2 * t + 1], s);
      sc[t] = s + madd;  // vector + scalar splat -> packed adds on WMMA tuples
    }
    // ---- online softmax per row; DPP ROW_XMASK butterflies ----
#pragma unroll
    for (int r = 0; r < 8; ++r) {
      float mx = fmaxf(fmaxf(sc[0][r], sc[1][r]), fmaxf(sc[2][r], sc[3][r]));
      mx = bfly_max16(mx);
      const float mnew = fmaxf(mrun[r], mx);
      const float al   = __builtin_amdgcn_exp2f(mrun[r] - mnew);
      mrun[r] = mnew;
      float p0 = __builtin_amdgcn_exp2f(sc[0][r] - mnew);
      float p1 = __builtin_amdgcn_exp2f(sc[1][r] - mnew);
      float p2 = __builtin_amdgcn_exp2f(sc[2][r] - mnew);
      float p3 = __builtin_amdgcn_exp2f(sc[3][r] - mnew);
      float ps = bfly_add16((p0 + p1) + (p2 + p3));
      lrun[r] = lrun[r] * al + ps;
      acc[0][r] *= al; acc[1][r] *= al; acc[2][r] *= al; acc[3][r] *= al;
      // C-layout -> per-wave LDS P tile
      __bf16* pr = P + (r + 8 * half) * PSTR + ln16;
      pr[0]  = (__bf16)p0;
      pr[16] = (__bf16)p1;
      pr[32] = (__bf16)p2;
      pr[48] = (__bf16)p3;
    }
    asm volatile("s_wait_dscnt 0x0" ::: "memory");
    // re-read P in A-operand layout (per-wave in-order DS makes this safe)
    bf16x16 pf0, pf1;
    {
      const __bf16* pp = P + (size_t)ln16 * PSTR + half * 8;
      pf0 = frag_ld2(pp,      pp + 16);   // keys kblk..kblk+31
      pf1 = frag_ld2(pp + 32, pp + 48);   // keys kblk+32..kblk+63
    }
    // ---- load ALL V fragments, then context += P @ V ----
    bf16x16 vf[8];
#pragma unroll
    for (int tj = 0; tj < 4; ++tj) {
      const __bf16* vp = &sV[buf][tj * 16 + ln16][half * 8];
      vf[2 * tj]     = frag_lds(vp);
      vf[2 * tj + 1] = frag_lds(vp + 32);
    }
#pragma unroll
    for (int tj = 0; tj < 4; ++tj) {
      f32x8 a = acc[tj];
      a = wmma_bf16(pf0, vf[2 * tj],     a);
      a = wmma_bf16(pf1, vf[2 * tj + 1], a);
      acc[tj] = a;
    }

    wait_async0();     // my prefetch copies have landed
    __syncthreads();   // everyone's copies landed, everyone done with `buf`
    buf ^= 1;
  }

  // ---- finalize: divide by l, write ctx [B*S, DM] bf16 ----
  float inv[8];
#pragma unroll
  for (int r = 0; r < 8; ++r) inv[r] = __builtin_amdgcn_rcpf(lrun[r]);
#pragma unroll
  for (int tj = 0; tj < 4; ++tj)
#pragma unroll
    for (int r = 0; r < 8; ++r)
      ctx[((size_t)b * SS + q0 + r + 8 * half) * DM + h * HD + tj * 16 + ln16] =
          (__bf16)(acc[tj][r] * inv[r]);
}

// ---------------- launch ----------------
extern "C" void kernel_launch(void* const* d_in, const int* in_sizes, int n_in,
                              void* d_out, int out_size, void* d_ws, size_t ws_size,
                              hipStream_t stream) {
  (void)in_sizes; (void)n_in; (void)out_size; (void)ws_size;
  const float* x    = (const float*)d_in[0];
  const int*   mask = (const int*)d_in[1];
  const float* Wq   = (const float*)d_in[2];
  const float* bq   = (const float*)d_in[3];
  const float* Wk   = (const float*)d_in[4];
  const float* bk   = (const float*)d_in[5];
  const float* Wv   = (const float*)d_in[6];
  const float* bv   = (const float*)d_in[7];
  const float* Wo   = (const float*)d_in[8];
  const float* bo   = (const float*)d_in[9];
  float* out = (float*)d_out;

  // workspace layout (48 MiB total)
  char* ws = (char*)d_ws;
  __bf16* xb  = (__bf16*)ws;                       // 4096*1024
  __bf16* wqb = xb  + (size_t)MROWS * DM;          // 1024*1024 each
  __bf16* wkb = wqb + (size_t)DM * DM;
  __bf16* wvb = wkb + (size_t)DM * DM;
  __bf16* wob = wvb + (size_t)DM * DM;
  __bf16* Qd  = wob + (size_t)DM * DM;             // [B,H,S,HD]
  __bf16* Kd  = Qd  + (size_t)MROWS * DM;
  __bf16* Vtd = Kd  + (size_t)MROWS * DM;          // [B,H,HD,S]
  __bf16* ctx = Vtd + (size_t)MROWS * DM;          // [B*S, DM]

  // fp32 -> bf16 (x and all weights)
  cvt_f32_bf16<<<(MROWS * DM) / 1024, 256, 0, stream>>>(x, xb);
  cvt_f32_bf16<<<(DM * DM) / 1024, 256, 0, stream>>>(Wq, wqb);
  cvt_f32_bf16<<<(DM * DM) / 1024, 256, 0, stream>>>(Wk, wkb);
  cvt_f32_bf16<<<(DM * DM) / 1024, 256, 0, stream>>>(Wv, wvb);
  cvt_f32_bf16<<<(DM * DM) / 1024, 256, 0, stream>>>(Wo, wob);

  const dim3 gg(DM / 128, MROWS / 128);  // (8, 32)
  proj_gemm<0><<<gg, 128, 0, stream>>>(xb, wqb, bq, Qd, nullptr);   // Q (scaled)
  proj_gemm<1><<<gg, 128, 0, stream>>>(xb, wkb, bk, Kd, nullptr);   // K
  proj_gemm<2><<<gg, 128, 0, stream>>>(xb, wvb, bv, Vtd, nullptr);  // V transposed

  attn_kernel<<<dim3(SS / 64, NH, BB), 128, 0, stream>>>(Qd, Kd, Vtd, mask, ctx);

  proj_gemm<3><<<gg, 128, 0, stream>>>(ctx, wob, bo, nullptr, out); // O -> f32 out
}
